// LayerwiseMLPUplift_65773129171678
// MI455X (gfx1250) — matrix operand order repaired and checked by reading, not compiled, not run on hardware
//
#include <hip/hip_runtime.h>
#include <hip/hip_bf16.h>
#include <math.h>

// ---------------------------------------------------------------------------
// Problem constants (fixed by the reference)
// ---------------------------------------------------------------------------
#define NTOK 8192
#define HDIM 768
#define IDIM 1536
#define LNUM 8
#define BM   64                   // token tile (uses CDNA5 320KB/WG LDS)
#define NPAD (NTOK + LNUM * BM)   // 8704 padded slots
#define NTILES (NPAD / BM)        // 136 tiles (fixed grid)

typedef unsigned short u16;
typedef __attribute__((ext_vector_type(16))) __bf16 v16bf;
typedef __attribute__((ext_vector_type(8)))  float  v8f;

struct alignas(16) Frag32B { uint4 lo, hi; };

// A-fragment (16x32 bf16): lanes 0-15 hold K0..7 & K16..23 of row M=lane,
// lanes 16-31 hold K8..15 & K24..31 (caller pre-adds the +8 lane offset).
__device__ __forceinline__ v16bf ld_a(const u16* p) {
  Frag32B t;
  t.lo = *(const uint4*)(p);
  t.hi = *(const uint4*)(p + 16);
  return __builtin_bit_cast(v16bf, t);
}
// B-fragment (32x16 bf16): same K order as A: lanes 0-15 hold K0..7 & K16..23,
// lanes 16-31 hold K8..15 & K24..31 (weights are pre-transposed, contiguous K).
__device__ __forceinline__ v16bf ld_b(const u16* p) {
  Frag32B t;
  t.lo = *(const uint4*)(p);
  t.hi = *(const uint4*)(p + 16);
  return __builtin_bit_cast(v16bf, t);
}
__device__ __forceinline__ u16 f2bf(float x) {
  __bf16 h = (__bf16)x;
  return __builtin_bit_cast(u16, h);
}
__device__ __forceinline__ float gelu_exact(float x) {
  return 0.5f * x * (1.0f + erff(x * 0.70710678118654752f));
}

// ---------------------------------------------------------------------------
// Routing: counts -> padded offsets -> slot assignment
// ---------------------------------------------------------------------------
__global__ void init_kernel(int* cnt, int* cur, int* perm) {
  int i = blockIdx.x * 256 + threadIdx.x;
  if (i < NPAD) perm[i] = -1;
  if (i < LNUM) { cnt[i] = 0; cur[i] = 0; }
}

__global__ void count_kernel(const int* __restrict__ lid, int* cnt) {
  int i = blockIdx.x * 256 + threadIdx.x;
  if (i < NTOK) atomicAdd(&cnt[lid[i]], 1);
}

__global__ void scan_kernel(const int* __restrict__ cnt, int* po) {
  if (threadIdx.x == 0 && blockIdx.x == 0) {
    int acc = 0;
    for (int e = 0; e < LNUM; ++e) {
      po[e] = acc;
      acc += (cnt[e] + BM - 1) & ~(BM - 1);  // pad each group to BM
    }
    po[LNUM] = acc;
  }
}

__global__ void assign_kernel(const int* __restrict__ lid, const int* __restrict__ po,
                              int* cur, int* perm) {
  int i = blockIdx.x * 256 + threadIdx.x;
  if (i < NTOK) {
    int e = lid[i];
    int s = atomicAdd(&cur[e], 1);
    perm[po[e] + s] = i;
  }
}

// Gather tokens in permuted order, convert to bf16 (pad rows -> 0).
__global__ void zconv_kernel(const float* __restrict__ z, const int* __restrict__ perm,
                             u16* __restrict__ zbf) {
  int slot = blockIdx.x;
  int tok = perm[slot];
  u16* d = zbf + (size_t)slot * HDIM;
  if (tok >= 0) {
    const float* s = z + (size_t)tok * HDIM;
    for (int c = threadIdx.x; c < HDIM; c += 256) d[c] = f2bf(s[c]);
  } else {
    for (int c = threadIdx.x; c < HDIM; c += 256) d[c] = 0;
  }
}

// dst[l][c][r] = bf16(src[l][r][c])  (R,C multiples of 32)
__global__ void transpose_cvt_kernel(const float* __restrict__ src, u16* __restrict__ dst,
                                     int R, int C) {
  __shared__ float tile[32][33];
  const int l = blockIdx.z;
  const int cb = blockIdx.x * 32, rb = blockIdx.y * 32;
  const float* s = src + (size_t)l * R * C;
  u16* d = dst + (size_t)l * R * C;
  const int tx = threadIdx.x, ty = threadIdx.y;
#pragma unroll
  for (int j = 0; j < 4; ++j)
    tile[ty * 4 + j][tx] = s[(size_t)(rb + ty * 4 + j) * C + cb + tx];
  __syncthreads();
#pragma unroll
  for (int j = 0; j < 4; ++j)
    d[(size_t)(cb + ty * 4 + j) * R + rb + tx] = f2bf(tile[tx][ty * 4 + j]);
}

// ---------------------------------------------------------------------------
// Fused expert MLP tile kernel: 64 tokens (one expert) per block, 16 waves
// arranged 4(M) x 4(N), each wave accumulating 4 N-subtiles (A reused 4x).
// GEMM1 (64x1536,K=768) -> GELU -> 192KB LDS bf16 act -> GEMM2 (64x768,K=1536)
// -> bias + residual scatter.  All matmuls on v_wmma_f32_16x16x32_bf16.
// ---------------------------------------------------------------------------
__global__ __launch_bounds__(512)
void fused_mlp_kernel(const u16* __restrict__ zbf, const int* __restrict__ perm,
                      const int* __restrict__ po, const u16* __restrict__ w1bf,
                      const u16* __restrict__ w2bf, const float* __restrict__ b1,
                      const float* __restrict__ b2, const float* __restrict__ z,
                      float* __restrict__ out) {
  extern __shared__ u16 act[];            // [BM][IDIM] bf16 = 192 KB
  __shared__ int stok[BM];

  const int rb = blockIdx.x * BM;
  if (rb >= po[LNUM]) return;             // uniform: empty padded tile
  int l = 0;
#pragma unroll
  for (int e = 1; e < LNUM; ++e) l = (rb >= po[e]) ? e : l;

  const int tid = threadIdx.x;
  if (tid < BM) stok[tid] = perm[rb + tid];
  __syncthreads();

  const int lane = tid & 31;
  const int wave = tid >> 5;              // 0..15
  const int wm   = wave & 3;              // wave M quarter (0..3)
  const int wn   = wave >> 2;             // wave N quarter (0..3)
  const int l15  = lane & 15;
  const int hi16 = (lane >> 4) & 1;
  const int aoff = hi16 * 8;              // A lane K offset per ISA layout
  const int boff = hi16 * 8;              // B lane K offset per ISA layout
  const int mrow = wm * 16 + hi16 * 8;    // C/D row base for this lane

  const u16* w1e = w1bf + (size_t)l * IDIM * HDIM;   // [i][h]
  const u16* w2e = w2bf + (size_t)l * HDIM * IDIM;   // [h][i]
  const u16* za  = zbf + (size_t)(rb + wm * 16 + l15) * HDIM + aoff;

  // ---------------- GEMM1 + bias + exact GELU -> act ----------------
  for (int nb = 0; nb < IDIM / 256; ++nb) {
    const int n0 = nb * 256 + wn * 64 + l15;          // 4 subtiles: n0+{0,16,32,48}
    const u16* p0 = w1e + (size_t)n0 * HDIM + boff;
    const u16* p1 = p0 + 16 * HDIM;
    const u16* p2 = p0 + 32 * HDIM;
    const u16* p3 = p0 + 48 * HDIM;
    v8f c0 = {}; v8f c1 = {}; v8f c2 = {}; v8f c3 = {};
    for (int kk = 0; kk < HDIM; kk += 32) {
      v16bf a  = ld_a(za + kk);
      v16bf b0 = ld_b(p0 + kk);
      v16bf b1f = ld_b(p1 + kk);
      v16bf b2f = ld_b(p2 + kk);
      v16bf b3f = ld_b(p3 + kk);
      c0 = __builtin_amdgcn_wmma_f32_16x16x32_bf16(false, a, false, b0,  (short)0, c0, false, false);
      c1 = __builtin_amdgcn_wmma_f32_16x16x32_bf16(false, a, false, b1f, (short)0, c1, false, false);
      c2 = __builtin_amdgcn_wmma_f32_16x16x32_bf16(false, a, false, b2f, (short)0, c2, false, false);
      c3 = __builtin_amdgcn_wmma_f32_16x16x32_bf16(false, a, false, b3f, (short)0, c3, false, false);
    }
    const float g0 = b1[l * IDIM + n0];
    const float g1 = b1[l * IDIM + n0 + 16];
    const float g2 = b1[l * IDIM + n0 + 32];
    const float g3 = b1[l * IDIM + n0 + 48];
#pragma unroll
    for (int r = 0; r < 8; ++r) {
      u16* arow = &act[(size_t)(mrow + r) * IDIM + n0];
      arow[0]  = f2bf(gelu_exact(c0[r] + g0));
      arow[16] = f2bf(gelu_exact(c1[r] + g1));
      arow[32] = f2bf(gelu_exact(c2[r] + g2));
      arow[48] = f2bf(gelu_exact(c3[r] + g3));
    }
  }
  __syncthreads();

  // ---------------- GEMM2 + bias + residual -> out ----------------
  const u16* aa = &act[(size_t)(wm * 16 + l15) * IDIM + aoff];
  for (int nb = 0; nb < HDIM / 256; ++nb) {
    const int n0 = nb * 256 + wn * 64 + l15;
    const u16* p0 = w2e + (size_t)n0 * IDIM + boff;
    const u16* p1 = p0 + 16 * IDIM;
    const u16* p2 = p0 + 32 * IDIM;
    const u16* p3 = p0 + 48 * IDIM;
    v8f c0 = {}; v8f c1 = {}; v8f c2 = {}; v8f c3 = {};
    for (int kk = 0; kk < IDIM; kk += 32) {
      v16bf a  = ld_a(aa + kk);                 // LDS activation tile
      v16bf b0 = ld_b(p0 + kk);
      v16bf b1f = ld_b(p1 + kk);
      v16bf b2f = ld_b(p2 + kk);
      v16bf b3f = ld_b(p3 + kk);
      c0 = __builtin_amdgcn_wmma_f32_16x16x32_bf16(false, a, false, b0,  (short)0, c0, false, false);
      c1 = __builtin_amdgcn_wmma_f32_16x16x32_bf16(false, a, false, b1f, (short)0, c1, false, false);
      c2 = __builtin_amdgcn_wmma_f32_16x16x32_bf16(false, a, false, b2f, (short)0, c2, false, false);
      c3 = __builtin_amdgcn_wmma_f32_16x16x32_bf16(false, a, false, b3f, (short)0, c3, false, false);
    }
    const float g0 = b2[l * HDIM + n0];
    const float g1 = b2[l * HDIM + n0 + 16];
    const float g2 = b2[l * HDIM + n0 + 32];
    const float g3 = b2[l * HDIM + n0 + 48];
#pragma unroll
    for (int r = 0; r < 8; ++r) {
      const int tok = stok[mrow + r];
      if (tok >= 0) {
        const size_t o = (size_t)tok * HDIM + n0;
        out[o]      = z[o]      + c0[r] + g0;
        out[o + 16] = z[o + 16] + c1[r] + g1;
        out[o + 32] = z[o + 32] + c2[r] + g2;
        out[o + 48] = z[o + 48] + c3[r] + g3;
      }
    }
  }
}

// ---------------------------------------------------------------------------
// Host launch.  Workspace layout (~51.3 MB required):
//   [0)      cnt[8]        [64)  cur[8]        [128) po[9]
//   [256)    perm[NPAD]                         (34816 B)
//   [35072)  zbf   bf16 [NPAD][H]               (13.37 MB)
//   [+)      w1bf  bf16 [L][I][H] (transposed)  (18.87 MB)
//   [+)      w2bf  bf16 [L][H][I] (transposed)  (18.87 MB)
// ---------------------------------------------------------------------------
extern "C" void kernel_launch(void* const* d_in, const int* in_sizes, int n_in,
                              void* d_out, int out_size, void* d_ws, size_t ws_size,
                              hipStream_t stream) {
  (void)in_sizes; (void)n_in; (void)out_size; (void)ws_size;
  const float* z   = (const float*)d_in[0];
  const int*   lid = (const int*)d_in[1];
  const float* w1  = (const float*)d_in[2];
  const float* b1  = (const float*)d_in[3];
  const float* w2  = (const float*)d_in[4];
  const float* b2  = (const float*)d_in[5];
  float* out = (float*)d_out;

  char* ws = (char*)d_ws;
  int* cnt  = (int*)(ws + 0);
  int* cur  = (int*)(ws + 64);
  int* po   = (int*)(ws + 128);
  int* perm = (int*)(ws + 256);
  size_t off = 256 + (size_t)NPAD * 4;                       // 35072, 256B aligned
  u16* zbf  = (u16*)(ws + off); off += (size_t)NPAD * HDIM * 2;
  u16* w1bf = (u16*)(ws + off); off += (size_t)LNUM * HDIM * IDIM * 2;
  u16* w2bf = (u16*)(ws + off);

  init_kernel  <<<(NPAD + 255) / 256, 256, 0, stream>>>(cnt, cur, perm);
  count_kernel <<<NTOK / 256, 256, 0, stream>>>(lid, cnt);
  scan_kernel  <<<1, 1, 0, stream>>>(cnt, po);
  assign_kernel<<<NTOK / 256, 256, 0, stream>>>(lid, po, cur, perm);
  zconv_kernel <<<NPAD, 256, 0, stream>>>(z, perm, zbf);

  dim3 tb(32, 8);
  transpose_cvt_kernel<<<dim3(IDIM / 32, HDIM / 32, LNUM), tb, 0, stream>>>(w1, w1bf, HDIM, IDIM);
  transpose_cvt_kernel<<<dim3(HDIM / 32, IDIM / 32, LNUM), tb, 0, stream>>>(w2, w2bf, IDIM, HDIM);

  fused_mlp_kernel<<<NTILES, 512, BM * IDIM * 2, stream>>>(
      zbf, perm, po, w1bf, w2bf, b1, b2, z, out);
}
